// ParallelDroplessMLP_27711128993793
// MI455X (gfx1250) — matrix operand order repaired and checked by reading, not compiled, not run on hardware
//
#include <hip/hip_runtime.h>
#include <math.h>

typedef __attribute__((ext_vector_type(16))) _Float16 v16h;
typedef __attribute__((ext_vector_type(8)))  float    v8f;

#define SLEN    2048
#define HS      512
#define FFN     2048
#define NE      8
#define TOPK    2
#define NTOK    (SLEN)              // BS = 1
#define TASSIGN (NTOK * TOPK)       // 4096 assignments
#define MAX_TILES (TASSIGN / 16 + NE)   // 264 worst case (per-expert padding)
#define FC_W    128                 // ffn chunk per iteration (8 waves x 16 cols)

// packed-weight geometry (B fragments: [tile][kchunk][lane][16 halves])
#define W1_NT   (FFN / 16)          // 128 n-tiles for GEMM1
#define W1_KC   (HS / 32)           // 16 k-chunks
#define W2_NT   (HS / 16)           // 32 n-tiles for GEMM2
#define W2_KC   (FFN / 32)          // 64 k-chunks
#define FRAG_H  512                 // halves per fragment (32 lanes x 16)
#define W1P_HALVES ((size_t)NE * W1_NT * W1_KC * FRAG_H)   // 8,388,608
#define W2P_HALVES ((size_t)NE * W2_NT * W2_KC * FRAG_H)   // 8,388,608
#define WS_TABLE_INTS 8192          // rowTok + tile tables, rounded up

__device__ __forceinline__ float gelu_exact(float v) {
    return 0.5f * v * (1.0f + erff(v * 0.70710678118654752440f));
}

// ---------------------------------------------------------------------------
// Kernel 1: stable bucket sort of assignments by expert + tile table build.
// ---------------------------------------------------------------------------
__global__ void moe_sort(const int* __restrict__ ei,
                         int* __restrict__ rowTok,
                         int* __restrict__ tE,
                         int* __restrict__ tS,
                         int* __restrict__ tR) {
    __shared__ int cnt[NE];
    __shared__ int off[NE + 1];
    const int t = threadIdx.x;
    if (t < NE) {
        int c = 0;
        for (int i = 0; i < TASSIGN; ++i) c += (ei[i] == t) ? 1 : 0;
        cnt[t] = c;
    }
    __syncthreads();
    if (t == 0) {
        off[0] = 0;
        for (int e = 0; e < NE; ++e) off[e + 1] = off[e] + cnt[e];
    }
    __syncthreads();
    if (t < NE) {
        int p = off[t];
        for (int i = 0; i < TASSIGN; ++i)
            if (ei[i] == t) rowTok[p++] = i >> 1;       // token = assignment/TOPK
    }
    if (t == 0) {
        int n = 0;
        for (int e = 0; e < NE; ++e)
            for (int s = off[e]; s < off[e + 1]; s += 16) {
                tE[n] = e; tS[n] = s;
                int rem = off[e + 1] - s;
                tR[n] = rem < 16 ? rem : 16;
                ++n;
            }
        for (; n < MAX_TILES; ++n) { tR[n] = 0; tE[n] = 0; tS[n] = 0; }
    }
}

// ---------------------------------------------------------------------------
// Packers: f32 weights -> f16 WMMA B-fragment order.
// B order chosen here (contiguous k, 32x16): n = lane&15 ; K = (lane>>4)*16 + h.
// frag element h = W[e][kc*32 + (lane>>4)*16 + h][16*nt + (lane&15)]
// ---------------------------------------------------------------------------
__global__ void pack_w1(const float* __restrict__ w1, _Float16* __restrict__ w1p) {
    const int gid = blockIdx.x * blockDim.x + threadIdx.x;
    if (gid >= NE * W1_NT * W1_KC * 32) return;
    const int lane = gid & 31;
    const int frag = gid >> 5;                 // ((e*W1_NT + nt)*W1_KC + kc)
    const int kc = frag % W1_KC;
    const int nt = (frag / W1_KC) % W1_NT;
    const int e  = frag / (W1_KC * W1_NT);
    const int g = lane >> 4, l16 = lane & 15;
    const float* src = w1 + ((size_t)e * HS + kc * 32 + g * 16) * FFN + 16 * nt + l16;
    v16h bF;
    #pragma unroll
    for (int h = 0; h < 16; ++h) bF[h] = (_Float16)src[(size_t)h * FFN];
    *(v16h*)(w1p + (size_t)frag * FRAG_H + lane * 16) = bF;
}

__global__ void pack_w2(const float* __restrict__ w2, _Float16* __restrict__ w2p) {
    const int gid = blockIdx.x * blockDim.x + threadIdx.x;
    if (gid >= NE * W2_NT * W2_KC * 32) return;
    const int lane = gid & 31;
    const int frag = gid >> 5;                 // ((e*W2_NT + nt)*W2_KC + kk)
    const int kk = frag % W2_KC;
    const int nt = (frag / W2_KC) % W2_NT;
    const int e  = frag / (W2_KC * W2_NT);
    const int g = lane >> 4, l16 = lane & 15;
    const float* src = w2 + ((size_t)e * FFN + kk * 32 + g * 16) * HS + 16 * nt + l16;
    v16h bF;
    #pragma unroll
    for (int h = 0; h < 16; ++h) bF[h] = (_Float16)src[(size_t)h * HS];
    *(v16h*)(w2p + (size_t)frag * FRAG_H + lane * 16) = bF;
}

// ---------------------------------------------------------------------------
// Kernel 2: fused grouped MLP. One block = 16 sorted rows x 512 out cols.
// PACKED=true: B fragments are single contiguous v16h loads from w1p/w2p.
// PACKED=false: fallback direct strided f32 loads (small-ws safety path).
// ---------------------------------------------------------------------------
template <bool PACKED>
__launch_bounds__(256, 1)
__global__ void moe_fused_mlp(const float* __restrict__ x,
                              const float* __restrict__ w1,
                              const float* __restrict__ w2,
                              const _Float16* __restrict__ w1p,
                              const _Float16* __restrict__ w2p,
                              const int* __restrict__ rowTok,
                              const int* __restrict__ tE,
                              const int* __restrict__ tS,
                              const int* __restrict__ tR,
                              float* __restrict__ buffer) {
    const int tile  = blockIdx.x;
    const int tRows = tR[tile];
    if (tRows == 0) return;                    // uniform scalar branch
    const int e  = tE[tile];
    const int s0 = tS[tile];

    const int tid  = threadIdx.x;
    const int wv   = tid >> 5;                 // wave 0..7
    const int lane = tid & 31;
    const int g    = lane >> 4;
    const int l16  = lane & 15;

    __shared__ _Float16 xRow[16][HS + 8];                          // x tile f16
    __shared__ __attribute__((aligned(32))) _Float16 hFrag[4][32][16]; // h in A-frag order

    // ---- stage x tile (gather sorted rows), f32 -> f16, zero-pad tail rows
    {
        const int row = tid >> 4;              // 0..15
        const int cb  = (tid & 15) * 32;       // 32 cols per thread
        const bool valid = (row < tRows);
        const int tok = valid ? rowTok[s0 + row] : 0;
        const float4* xp = (const float4*)(x + (size_t)tok * HS + cb);
        #pragma unroll
        for (int q = 0; q < 8; ++q) {
            float4 v = valid ? xp[q] : float4{0.f, 0.f, 0.f, 0.f};
            xRow[row][cb + 4 * q + 0] = (_Float16)v.x;
            xRow[row][cb + 4 * q + 1] = (_Float16)v.y;
            xRow[row][cb + 4 * q + 2] = (_Float16)v.z;
            xRow[row][cb + 4 * q + 3] = (_Float16)v.w;
        }
    }
    __syncthreads();

    // ---- persistent A fragments for GEMM1 (16 K-chunks of 32 over HS)
    // A order chosen here (contiguous k, 16x32): m=lane&15; K=g*16+h
    v16h aF[16];
    #pragma unroll
    for (int kc = 0; kc < 16; ++kc) {
        const int k0 = kc * 32;
        #pragma unroll
        for (int h = 0; h < 16; ++h) {
            const int kOff = g * 16 + h;
            aF[kc][h] = xRow[l16][k0 + kOff];
        }
    }

    v8f acc[4];                                // GEMM2 accum: cols 64*wv .. +63
    #pragma unroll
    for (int j = 0; j < 4; ++j) { v8f z = {}; acc[j] = z; }

    const float*    w1e  = w1 + (size_t)e * HS * FFN;
    const float*    w2e  = w2 + (size_t)e * FFN * HS;
    const _Float16* w1pE = w1p + (size_t)e * W1_NT * W1_KC * FRAG_H + lane * 16;
    const _Float16* w2pE = w2p + ((size_t)e * W2_NT + 4 * wv) * W2_KC * FRAG_H + lane * 16;

    // h producer -> hFrag placement (inverse of A layout):
    const int pkc2 = wv >> 1;                          // this wave's K-chunk of h
    const int ph   = l16;                              // half index
    _Float16* pBase = &hFrag[pkc2][(wv & 1) << 4][ph];   // + m*16 later

    for (int fc = 0; fc < FFN; fc += FC_W) {
        // ======== GEMM1: h(:, fc+16*wv : +16) = x_tile @ w1e, K = HS ========
        v8f hC = {};
        const int jn = (fc >> 4) + wv;                 // n-tile index
        const _Float16* pW1 = w1pE + (size_t)jn * W1_KC * FRAG_H;
        const int n1 = fc + 16 * wv + l16;             // fallback path col
        #pragma unroll
        for (int kc = 0; kc < 16; ++kc) {
            v16h bF;
            if (PACKED) {
                bF = *(const v16h*)(pW1 + (size_t)kc * FRAG_H);
            } else {
                const float* bp = w1e + (size_t)(kc * 32 + g * 16) * FFN + n1;
                #pragma unroll
                for (int h = 0; h < 16; ++h) bF[h] = (_Float16)bp[(size_t)h * FFN];
            }
            hC = __builtin_amdgcn_wmma_f32_16x16x32_f16(
                     false, aF[kc], false, bF, (short)0, hC, false, false);
        }

        // ---- exact GELU, write straight into A-fragment layout in LDS
        __syncthreads();                               // prior iter done reading
        #pragma unroll
        for (int r = 0; r < 8; ++r) {
            const int m = r + 8 * g;                   // C/D row
            pBase[m * 16] = (_Float16)gelu_exact(hC[r]);
        }
        __syncthreads();

        // ======== GEMM2: acc += h_chunk(16x128) @ w2e[fc:fc+128, :] ========
        #pragma unroll
        for (int kc2 = 0; kc2 < 4; ++kc2) {
            const v16h a2 = *(const v16h*)&hFrag[kc2][lane][0];
            const int kk = (fc >> 5) + kc2;            // global k-chunk of FFN
            #pragma unroll
            for (int j = 0; j < 4; ++j) {
                v16h bF;
                if (PACKED) {
                    bF = *(const v16h*)(w2pE + ((size_t)j * W2_KC + kk) * FRAG_H);
                } else {
                    const int n2 = 64 * wv + 16 * j + l16;
                    const float* bp = w2e + (size_t)(kk * 32 + g * 16) * HS + n2;
                    #pragma unroll
                    for (int h = 0; h < 16; ++h) bF[h] = (_Float16)bp[(size_t)h * HS];
                }
                acc[j] = __builtin_amdgcn_wmma_f32_16x16x32_f16(
                             false, a2, false, bF, (short)0, acc[j], false, false);
            }
        }

        // pull next iteration's w1 fragments toward the WGP (global_prefetch_b8)
        if (PACKED && fc + FC_W < FFN)
            __builtin_prefetch(pW1 + (size_t)8 * W1_KC * FRAG_H, 0, 1);
    }

    // ---- epilogue: buffer[token*NE + e][n] += acc (collisions add equal values)
    #pragma unroll
    for (int j = 0; j < 4; ++j) {
        const int n = 64 * wv + 16 * j + l16;
        #pragma unroll
        for (int r = 0; r < 8; ++r) {
            const int m = r + 8 * g;
            if (m < tRows) {
                const int tok = rowTok[s0 + m];
                atomicAdd(&buffer[((size_t)tok * NE + e) * HS + n], acc[j][r]);
            }
        }
    }
}

// ---------------------------------------------------------------------------
// Kernel 3: deterministic weighted combine y from buffer.
// ---------------------------------------------------------------------------
__global__ void moe_combine(const float* __restrict__ buffer,
                            const int* __restrict__ ei,
                            const float* __restrict__ ew,
                            float* __restrict__ y) {
    const int gid = blockIdx.x * blockDim.x + threadIdx.x;
    if (gid >= NTOK * HS) return;
    const int t = gid / HS;
    const int n = gid - t * HS;
    const int e0 = ei[2 * t], e1 = ei[2 * t + 1];
    const float wa = ew[2 * t], wb = ew[2 * t + 1];
    const float b0 = buffer[((size_t)t * NE + e0) * HS + n];
    const float b1 = buffer[((size_t)t * NE + e1) * HS + n];
    y[gid] = (e0 == e1) ? (wa + wb) * 0.5f * b0 : (wa * b0 + wb * b1);
}

// ---------------------------------------------------------------------------
extern "C" void kernel_launch(void* const* d_in, const int* in_sizes, int n_in,
                              void* d_out, int out_size, void* d_ws, size_t ws_size,
                              hipStream_t stream) {
    const float* x  = (const float*)d_in[0];
    const float* ew = (const float*)d_in[1];
    const int*   ei = (const int*)d_in[2];
    const float* w1 = (const float*)d_in[3];
    const float* w2 = (const float*)d_in[4];

    float* y      = (float*)d_out;                      // [NTOK, HS]
    float* buffer = (float*)d_out + (size_t)NTOK * HS;  // [NTOK, NE, HS]

    int* rowTok = (int*)d_ws;                           // [TASSIGN]
    int* tE = rowTok + TASSIGN;
    int* tS = tE + MAX_TILES;
    int* tR = tS + MAX_TILES;
    _Float16* w1p = (_Float16*)((char*)d_ws + (size_t)WS_TABLE_INTS * sizeof(int));
    _Float16* w2p = w1p + W1P_HALVES;

    const size_t wsNeed = (size_t)WS_TABLE_INTS * sizeof(int) +
                          (W1P_HALVES + W2P_HALVES) * sizeof(_Float16);   // ~33.6 MB
    const bool packed = ws_size >= wsNeed;

    hipMemsetAsync(d_out, 0, (size_t)out_size * sizeof(float), stream);
    moe_sort<<<1, 32, 0, stream>>>(ei, rowTok, tE, tS, tR);
    if (packed) {
        pack_w1<<<(NE * W1_NT * W1_KC * 32) / 256, 256, 0, stream>>>(w1, w1p);
        pack_w2<<<(NE * W2_NT * W2_KC * 32) / 256, 256, 0, stream>>>(w2, w2p);
        moe_fused_mlp<true><<<MAX_TILES, 256, 0, stream>>>(
            x, w1, w2, w1p, w2p, rowTok, tE, tS, tR, buffer);
    } else {
        moe_fused_mlp<false><<<MAX_TILES, 256, 0, stream>>>(
            x, w1, w2, w1p, w2p, rowTok, tE, tS, tR, buffer);
    }
    moe_combine<<<(NTOK * HS + 255) / 256, 256, 0, stream>>>(buffer, ei, ew, y);
}
